// HierarchicalAttention_52295521796645
// MI455X (gfx1250) — compile-verified, matrix operand order checked by reading, not run on hardware
//
#include <hip/hip_runtime.h>
#include <hip/hip_bf16.h>

typedef _Float16 f16_t;
typedef __attribute__((ext_vector_type(16))) _Float16 v16h;
typedef __attribute__((ext_vector_type(8)))  _Float16 v8h;
typedef __attribute__((ext_vector_type(8)))  float    v8f;

#define L_SEQ 6
#define E_DIM 512
#define QKV_N 3072      // 2 * 3 * E (local qkv | global qkv)
#define ATT_N 1024      // local out (512) | global out (512)

__device__ __constant__ float KMASK[36] = {
    1,1,1,1,1,1,
    1,1,0,1,0,0,
    1,0,1,1,0,0,
    1,1,1,1,1,1,
    1,0,0,1,1,0,
    1,0,0,1,0,1,
};

// ---------------------------------------------------------------------------
// CDNA5 async global->LDS copy (ASYNCcnt-tracked). Each lane copies 16B from
// its global address to its LDS byte-offset. Coalesced 512B per instruction.
// ---------------------------------------------------------------------------
__device__ __forceinline__ void async_copy16(unsigned int lds_off, const void* gptr) {
    unsigned long long ga = (unsigned long long)(uintptr_t)gptr;
    asm volatile("global_load_async_to_lds_b128 %0, %1, off"
                 :: "v"(lds_off), "v"(ga)
                 : "memory");
}
__device__ __forceinline__ void wait_async0() {
    asm volatile("s_wait_asynccnt 0x0" ::: "memory");
}
__device__ __forceinline__ unsigned int lds_off_of(const void* p) {
    return (unsigned int)(uintptr_t)p;   // LDS aperture: low 32 bits = LDS offset
}

// Load 16 consecutive fp32 (16B-aligned, global or LDS) -> 16xf16 WMMA fragment.
__device__ __forceinline__ v16h cvt_load16(const float* __restrict__ s) {
    const float4* s4 = (const float4*)s;
    float4 a = s4[0], b = s4[1], c = s4[2], d = s4[3];
    v16h r;
    r[0]=(_Float16)a.x;  r[1]=(_Float16)a.y;  r[2]=(_Float16)a.z;  r[3]=(_Float16)a.w;
    r[4]=(_Float16)b.x;  r[5]=(_Float16)b.y;  r[6]=(_Float16)b.z;  r[7]=(_Float16)b.w;
    r[8]=(_Float16)c.x;  r[9]=(_Float16)c.y;  r[10]=(_Float16)c.z; r[11]=(_Float16)c.w;
    r[12]=(_Float16)d.x; r[13]=(_Float16)d.y; r[14]=(_Float16)d.z; r[15]=(_Float16)d.w;
    return r;
}

// ---------------------------------------------------------------------------
// Kernel 1: Y[m, 0:3072] = X[m, 0:512] @ [w_in_l | w_in_g]^T + bias  (f16 out)
// 128x128 tile / block, 8 waves, each wave = 32x64 (2x4 WMMA 16x16x32).
// A/B tiles double-buffered in LDS via async copies; 144B row pitch is
// bank-conflict-free for the 16-row fragment reads (gcd(36,64)=4).
// ---------------------------------------------------------------------------
#define PITCH1 36                       // f32 per LDS row (32 data + 4 pad)
__global__ __launch_bounds__(256)
void qkv_gemm_kernel(const float* __restrict__ X,
                     const float* __restrict__ Wl,   // (1536, 512)
                     const float* __restrict__ Wg,   // (1536, 512)
                     const float* __restrict__ bl,
                     const float* __restrict__ bg,
                     f16_t* __restrict__ Y) {
    const int K  = E_DIM;          // 512
    const int Nl = 3 * E_DIM;      // 1536 (local|global weight boundary, 128-aligned)
    const int tileN = blockIdx.x * 128;
    const int tileM = blockIdx.y * 128;
    const int tid  = threadIdx.x;
    const int wave = tid >> 5, lane = tid & 31;
    const int lr = lane & 15, lk = lane >> 4;
    const int wm = wave & 3, wn = wave >> 2;

    __shared__ __align__(16) float sA[2][128 * PITCH1];
    __shared__ __align__(16) float sB[2][128 * PITCH1];
    const unsigned int aOff0 = lds_off_of(&sA[0][0]);
    const unsigned int bOff0 = lds_off_of(&sB[0][0]);
    const unsigned int bufBytes = 128 * PITCH1 * 4;

    const float* wbase = (tileN < Nl) ? (Wl + (size_t)tileN * K)
                                      : (Wg + (size_t)(tileN - Nl) * K);

    // stage k-step tiles: 128 rows x 32 f32 each (8 x 16B chunks / row)
    auto stage = [&](int k0, int buf) {
#pragma unroll
        for (int s = 0; s < 4; ++s) {
            const int chunk = tid + s * 256;          // 0..1023
            const int r = chunk >> 3, c = chunk & 7;
            async_copy16(aOff0 + buf * bufBytes + (r * PITCH1 + c * 4) * 4,
                         X + (size_t)(tileM + r) * K + k0 + c * 4);
            async_copy16(bOff0 + buf * bufBytes + (r * PITCH1 + c * 4) * 4,
                         wbase + (size_t)r * K + k0 + c * 4);
        }
    };

    v8f acc[2][4];
#pragma unroll
    for (int i = 0; i < 2; ++i)
#pragma unroll
        for (int j = 0; j < 4; ++j)
#pragma unroll
            for (int e = 0; e < 8; ++e) acc[i][j][e] = 0.0f;

    stage(0, 0);
    const int T = K / 32;                              // 16
    for (int t = 0; t < T; ++t) {
        wait_async0();                                 // my copies for step t done
        __syncthreads();                               // everyone's copies done,
                                                       // everyone done reading t-1
        if (t + 1 < T) stage((t + 1) * 32, (t + 1) & 1);

        const float* At = &sA[t & 1][0];
        const float* Bt = &sB[t & 1][0];
        v16h afrag[2], bfrag[4];
#pragma unroll
        for (int i = 0; i < 2; ++i)
            afrag[i] = cvt_load16(At + (wm * 32 + i * 16 + lr) * PITCH1 + lk * 16);
#pragma unroll
        for (int j = 0; j < 4; ++j)
            bfrag[j] = cvt_load16(Bt + (wn * 64 + j * 16 + lr) * PITCH1 + lk * 16);
#pragma unroll
        for (int i = 0; i < 2; ++i)
#pragma unroll
            for (int j = 0; j < 4; ++j)
                acc[i][j] = __builtin_amdgcn_wmma_f32_16x16x32_f16(
                    false, afrag[i], false, bfrag[j], (short)0, acc[i][j], false, false);
    }

    // Epilogue: C/D layout — lane lr = N, VGPR e -> M = e + 8*lk.
    const int rowBase = tileM + wm * 32;
    const int colBase = tileN + wn * 64;
#pragma unroll
    for (int j = 0; j < 4; ++j) {
        const int n = colBase + j * 16 + lr;
        const float bias = (n < Nl) ? bl[n] : bg[n - Nl];
#pragma unroll
        for (int i = 0; i < 2; ++i) {
#pragma unroll
            for (int e = 0; e < 8; ++e) {
                const int m = rowBase + i * 16 + lk * 8 + e;
                Y[(size_t)m * QKV_N + n] = (f16_t)(acc[i][j][e] + bias);
            }
        }
    }
}

// ---------------------------------------------------------------------------
// Kernel 2: per-(b, head) attention over L=6.
//   heads 0..3  : local  (hd=128, no mask)
//   heads 4..11 : global (hd=64, additive 0/1 float mask)
// ---------------------------------------------------------------------------
__global__ __launch_bounds__(256)
void attn_kernel(const f16_t* __restrict__ Y, f16_t* __restrict__ Z, int B) {
    const int b = blockIdx.x * blockDim.x + threadIdx.x;
    const int h = blockIdx.y;
    if (b >= B) return;

    const bool isLocal = (h < 4);
    const int hd = isLocal ? 128 : 64;
    int qcol, kcol, vcol, ocol;
    if (isLocal) {
        qcol = h * 128; kcol = 512 + h * 128; vcol = 1024 + h * 128; ocol = h * 128;
    } else {
        const int hg = h - 4;
        qcol = 1536 + hg * 64; kcol = 1536 + 512 + hg * 64;
        vcol = 1536 + 1024 + hg * 64; ocol = 512 + hg * 64;
    }
    const float scale = rsqrtf((float)hd);

    float s[L_SEQ][L_SEQ];
#pragma unroll
    for (int l = 0; l < L_SEQ; ++l)
#pragma unroll
        for (int m = 0; m < L_SEQ; ++m) s[l][m] = 0.0f;

    for (int dc = 0; dc < hd; dc += 8) {
        float q[L_SEQ][8], k[L_SEQ][8];
#pragma unroll
        for (int l = 0; l < L_SEQ; ++l) {
            v8h qh = *(const v8h*)(Y + (size_t)(l * B + b) * QKV_N + qcol + dc);
            v8h kh = *(const v8h*)(Y + (size_t)(l * B + b) * QKV_N + kcol + dc);
#pragma unroll
            for (int d = 0; d < 8; ++d) { q[l][d] = (float)qh[d]; k[l][d] = (float)kh[d]; }
        }
#pragma unroll
        for (int l = 0; l < L_SEQ; ++l)
#pragma unroll
            for (int m = 0; m < L_SEQ; ++m)
#pragma unroll
                for (int d = 0; d < 8; ++d) s[l][m] += q[l][d] * k[m][d];
    }

    float p[L_SEQ][L_SEQ];
#pragma unroll
    for (int l = 0; l < L_SEQ; ++l) {
        float mx = -1e30f;
#pragma unroll
        for (int m = 0; m < L_SEQ; ++m) {
            float v = s[l][m] * scale + (isLocal ? 0.0f : KMASK[l * 6 + m]);
            s[l][m] = v;
            mx = fmaxf(mx, v);
        }
        float sum = 0.0f;
#pragma unroll
        for (int m = 0; m < L_SEQ; ++m) { p[l][m] = __expf(s[l][m] - mx); sum += p[l][m]; }
        const float inv = 1.0f / sum;
#pragma unroll
        for (int m = 0; m < L_SEQ; ++m) p[l][m] *= inv;
    }

    for (int dc = 0; dc < hd; dc += 8) {
        float vv[L_SEQ][8];
#pragma unroll
        for (int m = 0; m < L_SEQ; ++m) {
            v8h vh = *(const v8h*)(Y + (size_t)(m * B + b) * QKV_N + vcol + dc);
#pragma unroll
            for (int d = 0; d < 8; ++d) vv[m][d] = (float)vh[d];
        }
#pragma unroll
        for (int l = 0; l < L_SEQ; ++l) {
            float o[8];
#pragma unroll
            for (int d = 0; d < 8; ++d) o[d] = 0.0f;
#pragma unroll
            for (int m = 0; m < L_SEQ; ++m)
#pragma unroll
                for (int d = 0; d < 8; ++d) o[d] += p[l][m] * vv[m][d];
            v8h oh;
#pragma unroll
            for (int d = 0; d < 8; ++d) oh[d] = (_Float16)o[d];
            *(v8h*)(Z + (size_t)(l * B + b) * ATT_N + ocol + dc) = oh;
        }
    }
}

// ---------------------------------------------------------------------------
// Kernel 3: Out[m, 0:512] = Z[m, 0:1024] @ [w_out_l | w_out_g]^T
//           + b_out_l + b_out_g     (fuses local_feat + global_feat add)
// A (f16) tile: 80B LDS pitch (gcd(20,64)=4 -> conflict-free v8h reads).
// ---------------------------------------------------------------------------
#define PITCH3A 40                      // f16 per LDS row (32 data + 8 pad)
__global__ __launch_bounds__(256)
void out_gemm_kernel(const f16_t* __restrict__ Z,
                     const float* __restrict__ Wl,   // (512, 512)
                     const float* __restrict__ Wg,   // (512, 512)
                     const float* __restrict__ bl,
                     const float* __restrict__ bg,
                     float* __restrict__ Out) {
    const int K  = ATT_N;          // 1024
    const int Kh = E_DIM;          // 512 boundary between w_out_l / w_out_g cols
    const int tileN = blockIdx.x * 128;
    const int tileM = blockIdx.y * 128;
    const int tid  = threadIdx.x;
    const int wave = tid >> 5, lane = tid & 31;
    const int lr = lane & 15, lk = lane >> 4;
    const int wm = wave & 3, wn = wave >> 2;

    __shared__ __align__(16) f16_t sA[2][128 * PITCH3A];
    __shared__ __align__(16) float sB[2][128 * PITCH1];
    const unsigned int aOff0 = lds_off_of(&sA[0][0]);
    const unsigned int bOff0 = lds_off_of(&sB[0][0]);
    const unsigned int aBufBytes = 128 * PITCH3A * 2;
    const unsigned int bBufBytes = 128 * PITCH1 * 4;

    auto stage = [&](int k0, int buf) {
        const float* wbase = (k0 < Kh) ? Wl : Wg;
        const int kk = (k0 < Kh) ? k0 : k0 - Kh;
        // A: 128 rows x 32 f16 = 4 x 16B chunks / row -> 512 chunks
#pragma unroll
        for (int s = 0; s < 2; ++s) {
            const int chunk = tid + s * 256;          // 0..511
            const int r = chunk >> 2, c = chunk & 3;
            async_copy16(aOff0 + buf * aBufBytes + (r * PITCH3A + c * 8) * 2,
                         Z + (size_t)(tileM + r) * K + k0 + c * 8);
        }
        // B: 128 rows x 32 f32 = 8 x 16B chunks / row -> 1024 chunks
#pragma unroll
        for (int s = 0; s < 4; ++s) {
            const int chunk = tid + s * 256;          // 0..1023
            const int r = chunk >> 3, c = chunk & 7;
            async_copy16(bOff0 + buf * bBufBytes + (r * PITCH1 + c * 4) * 4,
                         wbase + (size_t)(tileN + r) * Kh + kk + c * 4);
        }
    };

    v8f acc[2][4];
#pragma unroll
    for (int i = 0; i < 2; ++i)
#pragma unroll
        for (int j = 0; j < 4; ++j)
#pragma unroll
            for (int e = 0; e < 8; ++e) acc[i][j][e] = 0.0f;

    stage(0, 0);
    const int T = K / 32;                              // 32
    for (int t = 0; t < T; ++t) {
        wait_async0();
        __syncthreads();
        if (t + 1 < T) stage((t + 1) * 32, (t + 1) & 1);

        const f16_t* At = &sA[t & 1][0];
        const float* Bt = &sB[t & 1][0];
        v16h afrag[2], bfrag[4];
#pragma unroll
        for (int i = 0; i < 2; ++i) {
            const f16_t* ap = At + (wm * 32 + i * 16 + lr) * PITCH3A + lk * 16;
            v8h lo = *(const v8h*)ap;
            v8h hi = *(const v8h*)(ap + 8);
#pragma unroll
            for (int d = 0; d < 8; ++d) { afrag[i][d] = lo[d]; afrag[i][d + 8] = hi[d]; }
        }
#pragma unroll
        for (int j = 0; j < 4; ++j)
            bfrag[j] = cvt_load16(Bt + (wn * 64 + j * 16 + lr) * PITCH1 + lk * 16);
#pragma unroll
        for (int i = 0; i < 2; ++i)
#pragma unroll
            for (int j = 0; j < 4; ++j)
                acc[i][j] = __builtin_amdgcn_wmma_f32_16x16x32_f16(
                    false, afrag[i], false, bfrag[j], (short)0, acc[i][j], false, false);
    }

    const int rowBase = tileM + wm * 32;
    const int colBase = tileN + wn * 64;
#pragma unroll
    for (int j = 0; j < 4; ++j) {
        const int n = colBase + j * 16 + lr;
        const float bias = bl[n] + bg[n];
#pragma unroll
        for (int i = 0; i < 2; ++i) {
#pragma unroll
            for (int e = 0; e < 8; ++e) {
                const int m = rowBase + i * 16 + lk * 8 + e;
                Out[(size_t)m * E_DIM + n] = acc[i][j][e] + bias;
            }
        }
    }
}

extern "C" void kernel_launch(void* const* d_in, const int* in_sizes, int n_in,
                              void* d_out, int out_size, void* d_ws, size_t ws_size,
                              hipStream_t stream) {
    const float* x       = (const float*)d_in[0];
    const float* w_in_l  = (const float*)d_in[1];
    const float* b_in_l  = (const float*)d_in[2];
    const float* w_out_l = (const float*)d_in[3];
    const float* b_out_l = (const float*)d_in[4];
    const float* w_in_g  = (const float*)d_in[5];
    const float* b_in_g  = (const float*)d_in[6];
    const float* w_out_g = (const float*)d_in[7];
    const float* b_out_g = (const float*)d_in[8];

    const int B    = in_sizes[0] / (L_SEQ * E_DIM);   // 32768
    const int Mtot = L_SEQ * B;                        // 196608

    f16_t* Yqkv = (f16_t*)d_ws;                        // (Mtot, 3072) f16
    f16_t* Zatt = Yqkv + (size_t)Mtot * QKV_N;         // (Mtot, 1024) f16

    qkv_gemm_kernel<<<dim3(QKV_N / 128, Mtot / 128), 256, 0, stream>>>(
        x, w_in_l, w_in_g, b_in_l, b_in_g, Yqkv);

    attn_kernel<<<dim3((B + 255) / 256, 12), 256, 0, stream>>>(Yqkv, Zatt, B);

    out_gemm_kernel<<<dim3(E_DIM / 128, Mtot / 128), 256, 0, stream>>>(
        Zatt, w_out_l, w_out_g, b_out_l, b_out_g, (float*)d_out);
}
